// ABMIL_41266045780427
// MI455X (gfx1250) — matrix operand order repaired and checked by reading, not compiled, not run on hardware
//
#include <hip/hip_runtime.h>
#include <hip/hip_bf16.h>

typedef __bf16 bf16_t;
typedef bf16_t v16bf __attribute__((ext_vector_type(16)));
typedef float  v8f   __attribute__((ext_vector_type(8)));
typedef unsigned int u32x4 __attribute__((ext_vector_type(4)));

union FragU { u32x4 u[2]; v16bf v; };

#define R_ROWS 8200   // 8 * 1025
#define DDIM   1024
#define SLEN   1025
#define NB     8
#define NH     16
#define HD     64

// CDNA5 async global->LDS copy (VGLOBAL op 98, tracked by ASYNCcnt).
// lds_addr: per-lane LDS byte address; gptr: per-lane 64-bit global address.
__device__ __forceinline__ void async_copy_b128(unsigned lds_addr, const void* gptr) {
    asm volatile("global_load_async_to_lds_b128 %0, %1, off"
                 :: "v"(lds_addr), "v"(gptr) : "memory");
}
__device__ __forceinline__ void wait_async0() {
    asm volatile("s_wait_asynccnt 0x0" ::: "memory");
}

// ---------------------------------------------------------------- convert f32 -> bf16
__global__ __launch_bounds__(256) void cvt_kernel(const float* __restrict__ src,
                                                  bf16_t* __restrict__ dst, int n) {
    int i = blockIdx.x * 256 + threadIdx.x;
    if (i < n) dst[i] = (bf16_t)src[i];
}

// ---------------------------------------------------------------- concat(cls,x) + LayerNorm -> bf16
__global__ __launch_bounds__(256) void ln_kernel(const float* __restrict__ x,
                                                 const float* __restrict__ cls,
                                                 const float* __restrict__ gamma,
                                                 const float* __restrict__ beta,
                                                 bf16_t* __restrict__ lnb) {
    const int r = blockIdx.x;            // 0..8199
    const int b = r / SLEN;
    const int s = r % SLEN;
    const int tid = threadIdx.x;
    const float* src = (s == 0) ? cls : (x + ((size_t)b * 1024 + (s - 1)) * DDIM);

    __shared__ float red[256];

    float vals[4];
    float sum = 0.f;
#pragma unroll
    for (int i = 0; i < 4; i++) {
        vals[i] = src[tid + i * 256];
        sum += vals[i];
    }
    red[tid] = sum; __syncthreads();
    for (int o = 128; o > 0; o >>= 1) { if (tid < o) red[tid] += red[tid + o]; __syncthreads(); }
    const float mu = red[0] * (1.0f / DDIM);
    __syncthreads();

    float vs = 0.f;
#pragma unroll
    for (int i = 0; i < 4; i++) { float d = vals[i] - mu; vs += d * d; }
    red[tid] = vs; __syncthreads();
    for (int o = 128; o > 0; o >>= 1) { if (tid < o) red[tid] += red[tid + o]; __syncthreads(); }
    const float rstd = rsqrtf(red[0] * (1.0f / DDIM) + 1e-5f);

#pragma unroll
    for (int i = 0; i < 4; i++) {
        int d = tid + i * 256;
        lnb[(size_t)r * DDIM + d] = (bf16_t)((vals[i] - mu) * rstd * gamma[d] + beta[d]);
    }
}

// ---------------------------------------------------------------- K/V projections via bf16 WMMA
// Y[r, n] = sum_d ln[r, d] * W[n, d] + bias[n]
// Block tile 128x128, wave tile 32x64 (2x4 WMMA), k-step 32.
// Double-buffered LDS fed by GLOBAL_LOAD_ASYNC_TO_LDS_B128; one barrier per k-step.
__global__ __launch_bounds__(256) void gemm_kv_kernel(const bf16_t* __restrict__ lnb,
                                                      const bf16_t* __restrict__ wkb,
                                                      const bf16_t* __restrict__ wvb,
                                                      const float* __restrict__ bk,
                                                      const float* __restrict__ bv,
                                                      bf16_t* __restrict__ kout,
                                                      bf16_t* __restrict__ vout) {
    const int proj = blockIdx.z;
    const bf16_t* __restrict__ wb   = proj ? wvb : wkb;
    const float*  __restrict__ bias = proj ? bv  : bk;
    bf16_t*       __restrict__ out  = proj ? vout : kout;

    const int rowBase = blockIdx.x * 128;
    const int nBase   = blockIdx.y * 128;

    // ping-pong LDS tiles, padded row stride 40 bf16 (80B), 16B aligned
    __shared__ __align__(16) bf16_t As[2][128 * 40];
    __shared__ __align__(16) bf16_t Bs[2][128 * 40];

    const int tid   = threadIdx.x;
    const int lane  = tid & 31;
    const int w     = tid >> 5;
    const int waveM = w & 3;   // 4 waves in M, each owns 32 rows (2x16)
    const int waveN = w >> 2;  // 2 waves in N, each owns 64 cols (4x16)

    // per-thread staging coordinates (2 chunks of A, 2 of B per k-step)
    int aRow[2], aChunk[2], bRow[2], bChunk[2];
#pragma unroll
    for (int rep = 0; rep < 2; rep++) {
        int item = rep * 256 + tid;
        int i = item >> 2, c = item & 3;
        int row = rowBase + i; if (row >= R_ROWS) row = R_ROWS - 1;
        aRow[rep] = row;  aChunk[rep] = c;
        bRow[rep] = nBase + i; bChunk[rep] = c;
    }

    // issue async loads for one k-step into buffer `buf`
    auto stage = [&](int buf, int kb) {
#pragma unroll
        for (int rep = 0; rep < 2; rep++) {
            int item = rep * 256 + tid;
            int i = item >> 2;
            async_copy_b128((unsigned)(uintptr_t)&As[buf][i * 40 + aChunk[rep] * 8],
                            lnb + (size_t)aRow[rep] * DDIM + kb + aChunk[rep] * 8);
        }
#pragma unroll
        for (int rep = 0; rep < 2; rep++) {
            int item = rep * 256 + tid;
            int i = item >> 2;
            async_copy_b128((unsigned)(uintptr_t)&Bs[buf][i * 40 + bChunk[rep] * 8],
                            wb + (size_t)bRow[rep] * DDIM + kb + bChunk[rep] * 8);
        }
    };

    v8f acc[2][4];
#pragma unroll
    for (int mm = 0; mm < 2; mm++)
#pragma unroll
        for (int nn = 0; nn < 4; nn++)
#pragma unroll
            for (int j = 0; j < 8; j++) acc[mm][nn][j] = 0.f;

    const int m16  = lane & 15;    // matrix row/col within 16
    const int half = lane >> 4;    // lane half selects K group

    // compute one k-step from buffer `buf`
    auto compute = [&](int buf) {
        // A fragments (16x32): VGPR 0..3: K=half*8+0..7 ; VGPR 4..7: K=16+half*8+0..7
        FragU af[2];
#pragma unroll
        for (int mm = 0; mm < 2; mm++) {
            const bf16_t* base = &As[buf][(waveM * 32 + mm * 16 + m16) * 40 + half * 8];
            af[mm].u[0] = *(const u32x4*)(base);
            af[mm].u[1] = *(const u32x4*)(base + 16);
        }
        // B fragments (32x16): VGPR 0..3: K=half*8+0..7 ; VGPR 4..7: K=16+half*8+0..7
        FragU bfr[4];
#pragma unroll
        for (int nn = 0; nn < 4; nn++) {
            const bf16_t* base = &Bs[buf][(waveN * 64 + nn * 16 + m16) * 40 + half * 8];
            bfr[nn].u[0] = *(const u32x4*)(base);
            bfr[nn].u[1] = *(const u32x4*)(base + 16);
        }
#pragma unroll
        for (int mm = 0; mm < 2; mm++)
#pragma unroll
            for (int nn = 0; nn < 4; nn++)
                acc[mm][nn] = __builtin_amdgcn_wmma_f32_16x16x32_bf16(
                    false, af[mm].v, false, bfr[nn].v, (short)0, acc[mm][nn], false, false);
    };

    // prologue: fill buffer 0
    stage(0, 0);
    wait_async0();
    __syncthreads();

    // main loop, manually 2x unrolled so buffer indices are constant
    for (int kb = 0; kb < DDIM; kb += 64) {
        stage(1, kb + 32);            // prefetch next tile while computing
        compute(0);
        wait_async0();
        __syncthreads();

        if (kb + 64 < DDIM) stage(0, kb + 64);
        compute(1);
        if (kb + 64 < DDIM) { wait_async0(); __syncthreads(); }
    }

    // ---- epilogue: C/D layout lane l: n=l%16, m=(l/16)*8 + j
#pragma unroll
    for (int mm = 0; mm < 2; mm++)
#pragma unroll
        for (int nn = 0; nn < 4; nn++) {
            int col = nBase + waveN * 64 + nn * 16 + m16;
            float bcol = bias[col];
#pragma unroll
            for (int j = 0; j < 8; j++) {
                int row = rowBase + waveM * 32 + mm * 16 + half * 8 + j;
                if (row < R_ROWS) out[(size_t)row * DDIM + col] = (bf16_t)(acc[mm][nn][j] + bcol);
            }
        }
}

// ---------------------------------------------------------------- q = ln[b,0,:] @ Wq^T + bq (one wave per output)
__global__ __launch_bounds__(256) void qproj_kernel(const bf16_t* __restrict__ lnb,
                                                    const bf16_t* __restrict__ wqb,
                                                    const float* __restrict__ bq,
                                                    float* __restrict__ q) {
    const int gid  = blockIdx.x * 8 + (threadIdx.x >> 5);  // 0..8191
    const int lane = threadIdx.x & 31;
    const int b = gid >> 10;
    const int o = gid & 1023;
    const bf16_t* row  = lnb + (size_t)b * SLEN * DDIM;    // cls row
    const bf16_t* wrow = wqb + (size_t)o * DDIM;
    float a = 0.f;
    for (int d = lane; d < DDIM; d += 32)
        a += (float)row[d] * (float)wrow[d];
#pragma unroll
    for (int off = 16; off > 0; off >>= 1) a += __shfl_xor(a, off, 32);
    if (lane == 0) q[gid] = a + bq[o];
}

// ---------------------------------------------------------------- per-(b,h): scores -> softmax -> context
__global__ __launch_bounds__(256) void attn_kernel(const float* __restrict__ q,
                                                   const bf16_t* __restrict__ kb,
                                                   const bf16_t* __restrict__ vb,
                                                   float* __restrict__ ctx) {
    const int b = blockIdx.x >> 4;
    const int h = blockIdx.x & 15;
    const int tid = threadIdx.x;

    __shared__ float qs[HD];
    __shared__ float ps[SLEN];
    __shared__ float red[256];

    if (tid < HD) qs[tid] = q[b * DDIM + h * HD + tid];
    __syncthreads();

    float lmax = -3.4e38f;
    for (int s = tid; s < SLEN; s += 256) {
        const bf16_t* krow = kb + ((size_t)(b * SLEN + s)) * DDIM + h * HD;
        float a = 0.f;
#pragma unroll 8
        for (int d = 0; d < HD; d++) a += qs[d] * (float)krow[d];
        a *= 0.125f;                 // 1/sqrt(64)
        ps[s] = a;
        lmax = fmaxf(lmax, a);
    }
    red[tid] = lmax; __syncthreads();
    for (int o = 128; o > 0; o >>= 1) { if (tid < o) red[tid] = fmaxf(red[tid], red[tid + o]); __syncthreads(); }
    const float mx = red[0]; __syncthreads();

    float lsum = 0.f;
    for (int s = tid; s < SLEN; s += 256) {
        float e = __expf(ps[s] - mx);
        ps[s] = e;
        lsum += e;
    }
    red[tid] = lsum; __syncthreads();
    for (int o = 128; o > 0; o >>= 1) { if (tid < o) red[tid] += red[tid + o]; __syncthreads(); }
    const float inv = 1.f / red[0]; __syncthreads();

    // context: thread = (chunk c, dim d); coalesced 128B rows of V per step
    const int d = tid & 63;
    const int c = tid >> 6;
    float a = 0.f;
    for (int s = c; s < SLEN; s += 4)
        a += ps[s] * (float)vb[((size_t)(b * SLEN + s)) * DDIM + h * HD + d];
    red[tid] = a; __syncthreads();
    if (tid < HD) {
        float r = red[tid] + red[tid + 64] + red[tid + 128] + red[tid + 192];
        ctx[b * DDIM + h * HD + tid] = r * inv;
    }
}

// ---------------------------------------------------------------- out = ctx @ Wo^T + bo (one wave per output)
__global__ __launch_bounds__(256) void outproj_kernel(const float* __restrict__ ctx,
                                                      const bf16_t* __restrict__ wob,
                                                      const float* __restrict__ bo,
                                                      float* __restrict__ out) {
    const int gid  = blockIdx.x * 8 + (threadIdx.x >> 5);  // 0..8191
    const int lane = threadIdx.x & 31;
    const int b = gid >> 10;
    const int o = gid & 1023;
    const float*  crow = ctx + (size_t)b * DDIM;
    const bf16_t* wrow = wob + (size_t)o * DDIM;
    float a = 0.f;
    for (int d = lane; d < DDIM; d += 32)
        a += crow[d] * (float)wrow[d];
#pragma unroll
    for (int off = 16; off > 0; off >>= 1) a += __shfl_xor(a, off, 32);
    if (lane == 0) out[gid] = a + bo[o];
}

// ----------------------------------------------------------------
extern "C" void kernel_launch(void* const* d_in, const int* in_sizes, int n_in,
                              void* d_out, int out_size, void* d_ws, size_t ws_size,
                              hipStream_t stream) {
    const float* x     = (const float*)d_in[0];
    const float* cls   = (const float*)d_in[1];
    const float* gamma = (const float*)d_in[2];
    const float* beta  = (const float*)d_in[3];
    const float* w_q   = (const float*)d_in[4];
    const float* b_q   = (const float*)d_in[5];
    const float* w_k   = (const float*)d_in[6];
    const float* b_k   = (const float*)d_in[7];
    const float* w_v   = (const float*)d_in[8];
    const float* b_v   = (const float*)d_in[9];
    const float* w_o   = (const float*)d_in[10];
    const float* b_o   = (const float*)d_in[11];
    float* out = (float*)d_out;

    // workspace carve-up (256B aligned)
    char* ws = (char*)d_ws;
    size_t off = 0;
    auto carve = [&](size_t bytes) { size_t o = off; off = (off + bytes + 255) & ~(size_t)255; return o; };
    const size_t lnBytes = (size_t)R_ROWS * DDIM * sizeof(bf16_t);
    const size_t wBytes  = (size_t)DDIM * DDIM * sizeof(bf16_t);
    bf16_t* lnb = (bf16_t*)(ws + carve(lnBytes));
    bf16_t* wqb = (bf16_t*)(ws + carve(wBytes));
    bf16_t* wkb = (bf16_t*)(ws + carve(wBytes));
    bf16_t* wvb = (bf16_t*)(ws + carve(wBytes));
    bf16_t* wob = (bf16_t*)(ws + carve(wBytes));
    bf16_t* kbf = (bf16_t*)(ws + carve(lnBytes));
    bf16_t* vbf = (bf16_t*)(ws + carve(lnBytes));
    float*  qf  = (float*) (ws + carve((size_t)NB * DDIM * sizeof(float)));
    float*  cxf = (float*) (ws + carve((size_t)NB * DDIM * sizeof(float)));

    const int wN = DDIM * DDIM;
    const int cvtBlocks = (wN + 255) / 256;
    cvt_kernel<<<cvtBlocks, 256, 0, stream>>>(w_q, wqb, wN);
    cvt_kernel<<<cvtBlocks, 256, 0, stream>>>(w_k, wkb, wN);
    cvt_kernel<<<cvtBlocks, 256, 0, stream>>>(w_v, wvb, wN);
    cvt_kernel<<<cvtBlocks, 256, 0, stream>>>(w_o, wob, wN);

    ln_kernel<<<R_ROWS, 256, 0, stream>>>(x, cls, gamma, beta, lnb);

    dim3 g((R_ROWS + 127) / 128, DDIM / 128, 2);   // 65 x 8 x 2
    gemm_kv_kernel<<<g, 256, 0, stream>>>(lnb, wkb, wvb, b_k, b_v, kbf, vbf);

    qproj_kernel<<<NB * DDIM / 8, 256, 0, stream>>>(lnb, wqb, b_q, qf);

    attn_kernel<<<NB * NH, 256, 0, stream>>>(qf, kbf, vbf, cxf);

    outproj_kernel<<<NB * DDIM / 8, 256, 0, stream>>>(cxf, wob, b_o, out);
}
